// DeepSeekV3Router_54829552501187
// MI455X (gfx1250) — compile-verified, matrix operand order checked by reading, not run on hardware
//
#include <hip/hip_runtime.h>
#include <hip/hip_bf16.h>

#define T_TOK   8192
#define HIDDEN  7168
#define NEXP    256
#define KCHUNK  32
#define MTILE   64
#define ROW_DW  20   // padded LDS row stride in dwords (80 B) -> conflict-free b128 reads

typedef __attribute__((ext_vector_type(16))) __bf16 v16bf;
typedef __attribute__((ext_vector_type(8)))  float  v8f;
typedef __attribute__((ext_vector_type(4)))  float  v4f;   // native vector for NT loads
typedef __attribute__((ext_vector_type(2)))  float  v2f;
typedef __attribute__((ext_vector_type(2)))  __bf16 v2bf;

// f32x2 -> packed bf16x2 (RNE): vector fptrunc lowers to v_cvt_pk_bf16_f32
__device__ __forceinline__ unsigned pack2_bf16(float a, float b) {
  v2f vf = {a, b};
  union { v2bf v; unsigned u; } t;
  t.v = __builtin_convertvector(vf, v2bf);
  return t.u;
}
__device__ __forceinline__ unsigned short f2bf(float f) {
  union { __bf16 h; unsigned short u; } t;
  t.h = (__bf16)f;
  return t.u;
}

// ---------------- Kernel 1: W[D,E] f32 -> Wt[E,D] bf16 (tiled transpose) ----
__global__ __launch_bounds__(256) void convertW_kernel(const float* __restrict__ W,
                                                       unsigned short* __restrict__ Wt) {
  __shared__ float tile[64][65];
  const int e0 = blockIdx.x * 64, d0 = blockIdx.y * 64;
  const int tx = threadIdx.x & 63, ty = threadIdx.x >> 6; // ty in 0..3
#pragma unroll
  for (int i = 0; i < 16; ++i) {
    int dl = ty + 4 * i;
    tile[dl][tx] = W[(size_t)(d0 + dl) * NEXP + (e0 + tx)];
  }
  __syncthreads();
#pragma unroll
  for (int i = 0; i < 16; ++i) {
    int el = ty + 4 * i;
    Wt[(size_t)(e0 + el) * HIDDEN + (d0 + tx)] = f2bf(tile[tx][el]);
  }
}

// ---------------- WMMA fragment load: 16-bit A/B 16x32 (row-of-K layout) ----
// lanes 0-15: row, K0..7 (16B @+0) and K16..23 (16B @+32B)
// lanes 16-31: row, K8..15 (@+16B) and K24..31 (@+48B)
__device__ __forceinline__ v16bf load_frag16(const unsigned* lds, int rowBase, int lane) {
  const int r = rowBase + (lane & 15);
  const unsigned* p = lds + r * ROW_DW + ((lane >> 4) << 2);
  union { uint4 u[2]; v16bf v; } t;
  t.u[0] = *(const uint4*)(p);
  t.u[1] = *(const uint4*)(p + 8);
  return t.v;
}

// ---------------- Kernel 2: scores = sigmoid(x @ W) via bf16 WMMA ----------
// Double-buffered LDS; B staged via gfx1250 async global->LDS DMA; A staged
// through registers with the global-load latency hidden behind the WMMAs.
__global__ __launch_bounds__(256) void gemm_scores_kernel(const float* __restrict__ x,
                                                          const unsigned short* __restrict__ Wt,
                                                          float* __restrict__ scores) {
  __shared__ unsigned As[2][MTILE * ROW_DW];  // 2 x  5 KB
  __shared__ unsigned Bs[2][NEXP * ROW_DW];   // 2 x 20 KB

  const int t    = threadIdx.x;
  const int lane = t & 31, wid = t >> 5;
  const int tok0 = blockIdx.x * MTILE;
  const int m_base = (wid >> 2) * 32;  // 2 waves in M -> 2 m-tiles each
  const int n_base = (wid & 3) * 64;   // 4 waves in N -> 4 n-tiles each
  const int NKC = HIDDEN / KCHUNK;

  v8f acc[2][4] = {};

  // A staging: thread t stages 8 floats of row (t>>2), quarter (t&3)
  const int arow = t >> 2, aq = t & 3;
  const int aIdx = arow * ROW_DW + aq * 4;
  const float* aSrc = x + (size_t)(tok0 + arow) * HIDDEN + aq * 8;
  // B staging: thread t owns expert row n = t (32 bf16 = 64 B per chunk)
  const unsigned short* bSrc = Wt + (size_t)t * HIDDEN;

  auto loadA = [&](int kc, v4f& f0, v4f& f1) {
    const v4f* s4 = (const v4f*)(aSrc + kc * KCHUNK);
    f0 = __builtin_nontemporal_load(s4);      // x is streamed exactly once
    f1 = __builtin_nontemporal_load(s4 + 1);
  };
  auto storeA = [&](int buf, v4f f0, v4f f1) {
    uint4 p;
    p.x = pack2_bf16(f0.x, f0.y); p.y = pack2_bf16(f0.z, f0.w);
    p.z = pack2_bf16(f1.x, f1.y); p.w = pack2_bf16(f1.z, f1.w);
    *(uint4*)&As[buf][aIdx] = p;              // ds_store_b128 (addrspace(3) base)
  };
  auto stageB_async = [&](int kc, int buf) {
    const void* g = (const void*)(bSrc + kc * KCHUNK); // 64 B contiguous
    unsigned lds = (unsigned)(size_t)(const void*)&Bs[buf][t * ROW_DW];
    // INST_OFFSET applies to both LDS and global addresses (ISA async pseudocode)
    asm volatile(
        "global_load_async_to_lds_b128 %0, %1, off\n\t"
        "global_load_async_to_lds_b128 %0, %1, off offset:16\n\t"
        "global_load_async_to_lds_b128 %0, %1, off offset:32\n\t"
        "global_load_async_to_lds_b128 %0, %1, off offset:48"
        :: "v"(lds), "v"(g) : "memory");
  };

  // prologue: fill buffer 0
  {
    v4f f0, f1;
    loadA(0, f0, f1);
    stageB_async(0, 0);
    storeA(0, f0, f1);
    asm volatile("s_wait_asynccnt 0" ::: "memory");
    __syncthreads();
  }

  for (int kc = 0; kc < NKC; ++kc) {
    const int buf = kc & 1;
    const bool more = (kc + 1 < NKC);
    v4f nf0, nf1;
    if (more) {
      loadA(kc + 1, nf0, nf1);        // global loads issued now ...
      stageB_async(kc + 1, buf ^ 1);  // ... DMA runs in background
      if (kc + 2 < NKC)
        __builtin_prefetch(aSrc + (kc + 2) * KCHUNK, 0, 0); // global_prefetch_b8
    }

    v16bf a[2], b[4];
#pragma unroll
    for (int i = 0; i < 2; ++i) a[i] = load_frag16(&As[buf][0], m_base + 16 * i, lane);
#pragma unroll
    for (int j = 0; j < 4; ++j) b[j] = load_frag16(&Bs[buf][0], n_base + 16 * j, lane);

#pragma unroll
    for (int i = 0; i < 2; ++i)
#pragma unroll
      for (int j = 0; j < 4; ++j)
        acc[i][j] = __builtin_amdgcn_wmma_f32_16x16x32_bf16(
            false, a[i], false, b[j], (short)0, acc[i][j], false, false);

    if (more) storeA(buf ^ 1, nf0, nf1);  // load-wait lands after the WMMAs

    asm volatile("s_wait_asynccnt 0" ::: "memory");
    __syncthreads();
  }

  // --- epilogue: sigmoid, write scores[T,256] ---
  const int mrow_off = (lane >> 4) << 3; // VGPR r -> M = r (lanes 0-15) or r+8 (16-31)
  const int ncol = lane & 15;
#pragma unroll
  for (int i = 0; i < 2; ++i)
#pragma unroll
    for (int j = 0; j < 4; ++j)
#pragma unroll
      for (int r = 0; r < 8; ++r) {
        int m = m_base + 16 * i + r + mrow_off;
        int n = n_base + 16 * j + ncol;
        float v = acc[i][j][r];
        scores[(size_t)(tok0 + m) * NEXP + n] = 1.0f / (1.0f + __expf(-v));
      }
}

// ---------------- Kernel 3: grouped top-k routing (1 wave32 per token) -----
__global__ __launch_bounds__(256) void router_kernel(const float* __restrict__ scores,
                                                     const float* __restrict__ bias,
                                                     float* __restrict__ out) {
  const int lane = threadIdx.x & 31;
  const int wid  = threadIdx.x >> 5;
  const int tok  = blockIdx.x * 8 + wid;

  // each lane owns 8 consecutive experts: e = lane*8 + i ; group = lane>>2
  float s[8], sb[8];
  {
    const float4* sp = (const float4*)(scores + (size_t)tok * NEXP + lane * 8);
    float4 a0 = sp[0], a1 = sp[1];
    const float4* bp = (const float4*)(bias + lane * 8);
    float4 b0 = bp[0], b1 = bp[1];
    s[0]=a0.x; s[1]=a0.y; s[2]=a0.z; s[3]=a0.w;
    s[4]=a1.x; s[5]=a1.y; s[6]=a1.z; s[7]=a1.w;
    sb[0]=s[0]+b0.x; sb[1]=s[1]+b0.y; sb[2]=s[2]+b0.z; sb[3]=s[3]+b0.w;
    sb[4]=s[4]+b1.x; sb[5]=s[5]+b1.y; sb[6]=s[6]+b1.z; sb[7]=s[7]+b1.w;
  }

  // per-lane top-2 of biased scores
  float m1 = -1e30f, m2 = -1e30f;
#pragma unroll
  for (int i = 0; i < 8; ++i) {
    float v = sb[i];
    if (v > m1) { m2 = m1; m1 = v; } else if (v > m2) { m2 = v; }
  }
  // merge top-2 across the 4 lanes of the group (xor 1, 2)
#pragma unroll
  for (int off = 1; off <= 2; off <<= 1) {
    float o1 = __shfl_xor(m1, off, 32);
    float o2 = __shfl_xor(m2, off, 32);
    float nm1 = fmaxf(m1, o1);
    float nm2 = fmaxf(fminf(m1, o1), fmaxf(m2, o2));
    m1 = nm1; m2 = nm2;
  }
  const float gscore = m1 + m2;
  const int myg = lane >> 2;
  // rank my group among 8 (ties: lower group index wins)
  int rank = 0;
#pragma unroll
  for (int g = 0; g < 8; ++g) {
    float sg = __shfl(gscore, g * 4, 32);
    rank += (sg > gscore) || (sg == gscore && g < myg);
  }
  const bool gsel = rank < 4;
  float sm[8];
#pragma unroll
  for (int i = 0; i < 8; ++i) sm[i] = gsel ? sb[i] : 0.0f;

  // 8 argmax rounds over 256 experts
  unsigned pickedMask = 0;
  float wsum = 0.0f, out_w = 0.0f;
  int out_idx = 0;
#pragma unroll
  for (int r = 0; r < 8; ++r) {
    float bv = -1e30f, bs = 0.0f;
    int bi = 0x7FFFFFFF;
#pragma unroll
    for (int i = 0; i < 8; ++i) {
      bool avail = !((pickedMask >> i) & 1u);
      if (avail && sm[i] > bv) { bv = sm[i]; bi = lane * 8 + i; bs = s[i]; }
    }
#pragma unroll
    for (int off = 16; off >= 1; off >>= 1) {
      float ov = __shfl_xor(bv, off, 32);
      int   oi = __shfl_xor(bi, off, 32);
      float os = __shfl_xor(bs, off, 32);
      if (ov > bv || (ov == bv && oi < bi)) { bv = ov; bi = oi; bs = os; }
    }
    if ((bi >> 3) == lane) pickedMask |= 1u << (bi & 7);
    wsum += bs;
    if (lane == r) { out_idx = bi; out_w = bs; }
  }

  const float scale = 2.5f / (wsum + 1e-20f);
  if (lane < 8) {
    out[(size_t)tok * 8 + lane] = out_w * scale;                       // weights
    out[(size_t)T_TOK * 8 + (size_t)tok * 8 + lane] = (float)out_idx;  // indices
  }
}

// ---------------- host launch ----------------------------------------------
extern "C" void kernel_launch(void* const* d_in, const int* in_sizes, int n_in,
                              void* d_out, int out_size, void* d_ws, size_t ws_size,
                              hipStream_t stream) {
  const float* x    = (const float*)d_in[0];   // [T, HIDDEN]
  const float* W    = (const float*)d_in[1];   // [HIDDEN, NEXP]
  const float* bias = (const float*)d_in[2];   // [NEXP]
  float* out = (float*)d_out;                  // [T*8 weights][T*8 indices]

  unsigned short* Wt = (unsigned short*)d_ws;                        // 3.67 MB bf16
  float* scores = (float*)((char*)d_ws + (size_t)NEXP * HIDDEN * 2); // 8 MB f32

  dim3 cg(NEXP / 64, HIDDEN / 64);
  convertW_kernel<<<cg, 256, 0, stream>>>(W, Wt);
  gemm_scores_kernel<<<T_TOK / MTILE, 256, 0, stream>>>(x, Wt, scores);
  router_kernel<<<T_TOK / 8, 256, 0, stream>>>(scores, bias, out);
}
